// Model_29351806501636
// MI455X (gfx1250) — matrix operand order repaired and checked by reading, not compiled, not run on hardware
//
#include <hip/hip_runtime.h>
#include <cmath>

// Model dims (match reference)
#define V_SZ   30000
#define D_SZ   256
#define H_SZ   4
#define DK_SZ  64
#define N_SZ   1024
#define B_SZ   16
#define HID_SZ 256

typedef __attribute__((ext_vector_type(16))) _Float16 v16h;
typedef __attribute__((ext_vector_type(8)))  _Float16 v8h;
typedef __attribute__((ext_vector_type(8)))  float    v8f;

// ---------------------------------------------------------------------------
// WMMA fragment layouts per CDNA5 ISA 7.12.2 (wave32):
//  A (16x32 f16, MxK): lane L -> row m = L&15, half hi = L>>4.
//    elem pairs (2r,2r+1) <-> K = (r<4 ? 2r : 16+2(r-4)) + 8*hi
//    => in f16 memory: halves [base .. base+7] and [base+16 .. base+23],
//       base = rowptr + k0 + 8*hi. Two 16B vector loads.
//  B (32x16 f16, KxN): lane L -> col n = L&15; same K order as A
//    => halves [base .. base+7] and [base+16 .. base+23]. Two 16B vector loads.
//  C/D (16x16 f32): lane L -> col n = L&15 of rows r + 8*hi.
// ---------------------------------------------------------------------------
__device__ __forceinline__ v16h load_a_frag_h(const _Float16* __restrict__ p) {
    // p = row base + k0 + 8*hi (16B aligned)
    const v8h* pv = (const v8h*)p;
    const v8h lo = pv[0];   // K offsets 0..7
    const v8h hh = pv[2];   // K offsets 16..23
    v16h a;
#pragma unroll
    for (int i = 0; i < 8; ++i) { a[i] = lo[i]; a[8 + i] = hh[i]; }
    return a;
}

__device__ __forceinline__ v16h load_b_frag_h(const _Float16* __restrict__ p) {
    // p = source row base + k0 + 8*hi (16B aligned), halves 0..7 and 16..23
    const v8h* pv = (const v8h*)p;
    const v8h lo = pv[0];
    const v8h hh = pv[2];
    v16h b;
#pragma unroll
    for (int i = 0; i < 8; ++i) { b[i] = lo[i]; b[8 + i] = hh[i]; }
    return b;
}

// ---------------------------------------------------------------------------
// f32 -> f16 conversion (weights / one-time copies)
// ---------------------------------------------------------------------------
__global__ void cvt_h_kernel(const float* __restrict__ src, _Float16* __restrict__ dst, int n) {
    const int i = blockIdx.x * blockDim.x + threadIdx.x;
    if (i < n) dst[i] = (_Float16)src[i];
}

// ---------------------------------------------------------------------------
// GEMM: out[M,N] = A[M,K](f16) @ W[N,K](f16)^T + bias (+residual f32) (opt GELU)
// outF (f32) and/or outH (f16) may be null. block = 128 (4 waves, one 16x16
// tile each); grid = (M/16, N/64).
// ---------------------------------------------------------------------------
__global__ void gemm_h_kernel(const _Float16* __restrict__ A, const _Float16* __restrict__ W,
                              const float* __restrict__ bias, const float* __restrict__ residual,
                              float* __restrict__ outF, _Float16* __restrict__ outH,
                              int M, int N, int K, int act) {
    const int lane = threadIdx.x & 31;
    const int wave = threadIdx.x >> 5;
    const int m0 = blockIdx.x * 16;
    const int n0 = blockIdx.y * 64 + wave * 16;
    if (m0 >= M || n0 >= N) return;
    const int hi  = lane >> 4;
    const int l15 = lane & 15;
    const _Float16* arow = A + (size_t)(m0 + l15) * K + 8 * hi;
    const _Float16* wrow = W + (size_t)(n0 + l15) * K + 8 * hi;

    v8f c = {};
    for (int k0 = 0; k0 < K; k0 += 32) {
        v16h a = load_a_frag_h(arow + k0);
        v16h b = load_b_frag_h(wrow + k0);
        c = __builtin_amdgcn_wmma_f32_16x16x32_f16(false, a, false, b,
                                                   (short)0, c, false, false);
    }
#pragma unroll
    for (int r = 0; r < 8; ++r) {
        const int row = m0 + r + 8 * hi;
        const int col = n0 + l15;
        float val = c[r] + bias[col];
        if (residual) val += residual[(size_t)row * N + col];
        if (act == 1)  // exact GELU: 0.5*x*(1+erf(x/sqrt(2)))
            val = 0.5f * val * (1.0f + erff(val * 0.70710678118654752f));
        if (outF) outF[(size_t)row * N + col] = val;
        if (outH) outH[(size_t)row * N + col] = (_Float16)val;
    }
}

// ---------------------------------------------------------------------------
// Fused masked attention (flash-style), f16 Q/K/V. One wave per
// (batch, head, 16-query tile). Online softmax; P transposed via LDS.
// grid = B*H*(N/16) = 4096 blocks, block = 32.
// ---------------------------------------------------------------------------
__global__ void attention_kernel(const _Float16* __restrict__ q, const _Float16* __restrict__ k,
                                 const _Float16* __restrict__ v, const int* __restrict__ spans,
                                 _Float16* __restrict__ att) {
    __shared__ float pl[16][33];  // 16x32 P tile, padded row stride

    const int lane = threadIdx.x & 31;
    const int hi   = lane >> 4;
    const int l15  = lane & 15;
    const int bid  = blockIdx.x;
    const int tile = bid & 63;         // N/16 = 64 tiles
    const int h    = (bid >> 6) & 3;   // H = 4
    const int b    = bid >> 8;
    const int start = spans[2 * b];
    const int endi  = spans[2 * b + 1];
    const int hc   = h * DK_SZ;

    // Q fragments for head-dim chunks [0,32) and [32,64)
    const _Float16* qrow = q + (size_t)(b * N_SZ + tile * 16 + l15) * D_SZ + hc + 8 * hi;
    const v16h aq0 = load_a_frag_h(qrow);
    const v16h aq1 = load_a_frag_h(qrow + 32);

    bool qin[8];
#pragma unroll
    for (int r = 0; r < 8; ++r) {
        const int qn = tile * 16 + r + 8 * hi;
        qin[r] = (qn >= start) && (qn < endi);
    }

    float rmax[8], rsum[8];
#pragma unroll
    for (int r = 0; r < 8; ++r) { rmax[r] = -3.0e38f; rsum[r] = 0.0f; }
    v8f o[4] = {};

    for (int kb = 0; kb < N_SZ; kb += 32) {
        // ---- S = scale * Q K^T + mask, for 32 keys -----------------------
        v8f s[2] = {};
#pragma unroll
        for (int t = 0; t < 2; ++t) {
            const int key = kb + 16 * t + l15;
            const _Float16* krow = k + (size_t)(b * N_SZ + key) * D_SZ + hc + 8 * hi;
            v16h b0 = load_b_frag_h(krow);
            s[t] = __builtin_amdgcn_wmma_f32_16x16x32_f16(false, aq0, false, b0,
                                                          (short)0, s[t], false, false);
            v16h b1 = load_b_frag_h(krow + 32);
            s[t] = __builtin_amdgcn_wmma_f32_16x16x32_f16(false, aq1, false, b1,
                                                          (short)0, s[t], false, false);
        }
#pragma unroll
        for (int t = 0; t < 2; ++t) {
            const int key = kb + 16 * t + l15;
            const bool kin = (key >= start) && (key < endi);
#pragma unroll
            for (int r = 0; r < 8; ++r) {
                float val = s[t][r] * 0.125f;  // 1/sqrt(64)
                if (!(qin[r] && kin)) val += -1.0e10f;
                s[t][r] = val;
            }
        }
        // ---- online softmax (rows live in one wave-half: shfl masks < 16) -
        float p0[8], p1[8];
#pragma unroll
        for (int r = 0; r < 8; ++r) {
            float m = fmaxf(s[0][r], s[1][r]);
            m = fmaxf(m, __shfl_xor(m, 8, 32));
            m = fmaxf(m, __shfl_xor(m, 4, 32));
            m = fmaxf(m, __shfl_xor(m, 2, 32));
            m = fmaxf(m, __shfl_xor(m, 1, 32));
            const float mnew = fmaxf(rmax[r], m);
            const float corr = expf(rmax[r] - mnew);
            const float e0 = expf(s[0][r] - mnew);
            const float e1 = expf(s[1][r] - mnew);
            float ssum = e0 + e1;
            ssum += __shfl_xor(ssum, 8, 32);
            ssum += __shfl_xor(ssum, 4, 32);
            ssum += __shfl_xor(ssum, 2, 32);
            ssum += __shfl_xor(ssum, 1, 32);
            rsum[r] = rsum[r] * corr + ssum;
            rmax[r] = mnew;
            p0[r] = e0; p1[r] = e1;
#pragma unroll
            for (int dt = 0; dt < 4; ++dt) o[dt][r] *= corr;
        }
        // ---- transpose P (C layout -> A layout) through LDS ---------------
        __syncthreads();
#pragma unroll
        for (int r = 0; r < 8; ++r) {
            pl[r + 8 * hi][l15]      = p0[r];
            pl[r + 8 * hi][16 + l15] = p1[r];
        }
        __syncthreads();
        v16h pa;
        {
            const float* prow = &pl[l15][8 * hi];
#pragma unroll
            for (int r = 0; r < 8; ++r) {
                const int kk = (r < 4) ? (2 * r) : (16 + 2 * (r - 4));
                pa[2 * r]     = (_Float16)prow[kk];
                pa[2 * r + 1] = (_Float16)prow[kk + 1];
            }
        }
        // ---- O += P @ V (B fragment strided: K = key index) ---------------
#pragma unroll
        for (int dt = 0; dt < 4; ++dt) {
            v16h bv;
#pragma unroll
            for (int i = 0; i < 16; ++i) {
                const int key = kb + ((i < 8) ? (8 * hi + i) : (16 + 8 * hi + (i - 8)));
                bv[i] = v[(size_t)(b * N_SZ + key) * D_SZ + hc + 16 * dt + l15];
            }
            o[dt] = __builtin_amdgcn_wmma_f32_16x16x32_f16(false, pa, false, bv,
                                                           (short)0, o[dt], false, false);
        }
    }
    // ---- normalize + store (heads concatenated along D) -------------------
#pragma unroll
    for (int r = 0; r < 8; ++r) {
        const int row = b * N_SZ + tile * 16 + r + 8 * hi;
        const float inv = 1.0f / rsum[r];
#pragma unroll
        for (int dt = 0; dt < 4; ++dt)
            att[(size_t)row * D_SZ + hc + 16 * dt + l15] = (_Float16)(o[dt][r] * inv);
    }
}

// ---------------------------------------------------------------------------
// Embedding gather + positional encoding; writes f32 master + f16 GEMM copy.
// ---------------------------------------------------------------------------
__global__ void embed_kernel(const int* __restrict__ ids, const float* __restrict__ embW,
                             const float* __restrict__ embB, float* __restrict__ x,
                             _Float16* __restrict__ xh) {
    const int idx = blockIdx.x * blockDim.x + threadIdx.x;
    if (idx >= B_SZ * N_SZ * D_SZ) return;
    const int d  = idx & (D_SZ - 1);
    const int bn = idx >> 8;             // b*N + n
    const int n  = bn & (N_SZ - 1);
    const int tok = ids[bn];
    float val = embW[(size_t)d * V_SZ + tok] + embB[d];
    const int j2 = d & ~1;
    const float freq = expf(-(float)j2 * (9.210340371976184f / (float)D_SZ)); // ln(1e4)/D
    const float ang  = (float)n * freq;
    val += ((d & 1) ? cosf(ang) : sinf(ang)) * 0.1f;
    x[idx]  = val;
    xh[idx] = (_Float16)val;
}

// spans per batch: start = first true (else 0), end = LAST true index (else N-1)
// -> in_span(i) = (i >= start) && (i < end)   [matches reference's argmax quirk]
__global__ void spans_kernel(const int* __restrict__ mask, int* __restrict__ spans) {
    const int b = threadIdx.x;
    if (b >= B_SZ) return;
    const int* m = mask + b * N_SZ;
    int first = -1, last = -1;
    for (int i = 0; i < N_SZ; ++i)
        if (m[i] != 0) { if (first < 0) first = i; last = i; }
    spans[2 * b]     = (first >= 0) ? first : 0;
    spans[2 * b + 1] = (last  >= 0) ? last  : (N_SZ - 1);
}

__global__ void pool_kernel(const float* __restrict__ x, float* __restrict__ pooled) {
    const int idx = blockIdx.x * blockDim.x + threadIdx.x;  // B*D
    if (idx >= B_SZ * D_SZ) return;
    const int b = idx >> 8, d = idx & (D_SZ - 1);
    const float* p = x + (size_t)b * N_SZ * D_SZ + d;
    float s = 0.0f;
    for (int n = 0; n < N_SZ; ++n) s += p[(size_t)n * D_SZ];
    pooled[idx] = s * (1.0f / (float)N_SZ);
}

// MLP head: h = relu(BN(pooled @ h1_w^T + b1)); out = h @ h2_w^T + b2
__global__ void head_kernel(const float* __restrict__ pooled,
                            const float* __restrict__ h1w, const float* __restrict__ h1b,
                            const float* __restrict__ g,  const float* __restrict__ be,
                            const float* __restrict__ mu, const float* __restrict__ var,
                            const float* __restrict__ h2w, const float* __restrict__ h2b,
                            float* __restrict__ out) {
    __shared__ float hb[128];
    const int b = blockIdx.x, t = threadIdx.x;
    const float* p = pooled + b * D_SZ;
    const float* w = h1w + t * D_SZ;
    float s = h1b[t];
    for (int d = 0; d < D_SZ; ++d) s += p[d] * w[d];
    s = (s - mu[t]) * rsqrtf(var[t] + 1e-5f) * g[t] + be[t];
    hb[t] = fmaxf(s, 0.0f);
    __syncthreads();
    if (t == 0) {
        float o = h2b[0];
        for (int j = 0; j < 128; ++j) o += hb[j] * h2w[j];
        out[b] = o;
    }
}

// ---------------------------------------------------------------------------
extern "C" void kernel_launch(void* const* d_in, const int* in_sizes, int n_in,
                              void* d_out, int out_size, void* d_ws, size_t ws_size,
                              hipStream_t stream) {
    const int*   ids   = (const int*)d_in[0];
    const int*   amask = (const int*)d_in[1];
    const float* embW  = (const float*)d_in[2];
    const float* embB  = (const float*)d_in[3];
    const float* wq    = (const float*)d_in[4];
    const float* bq    = (const float*)d_in[5];
    const float* wk    = (const float*)d_in[6];
    const float* bk    = (const float*)d_in[7];
    const float* wv    = (const float*)d_in[8];
    const float* bv    = (const float*)d_in[9];
    const float* fcw   = (const float*)d_in[10];
    const float* fcb   = (const float*)d_in[11];
    const float* ff1w  = (const float*)d_in[12];
    const float* ff1b  = (const float*)d_in[13];
    const float* ff2w  = (const float*)d_in[14];
    const float* ff2b  = (const float*)d_in[15];
    const float* h1w   = (const float*)d_in[16];
    const float* h1b   = (const float*)d_in[17];
    const float* bng   = (const float*)d_in[18];
    const float* bnb   = (const float*)d_in[19];
    const float* bnm   = (const float*)d_in[20];
    const float* bnv   = (const float*)d_in[21];
    const float* h2w   = (const float*)d_in[22];
    const float* h2b   = (const float*)d_in[23];

    const size_t BUF = (size_t)B_SZ * N_SZ * D_SZ;  // 4,194,304 elements
    const int    WN  = D_SZ * D_SZ;                 // 65,536 weight elements

    // f32 region
    float* ws     = (float*)d_ws;
    float* x      = ws;                 // embeddings (residual stream), 16 MB
    float* x3     = x;                  // FFN2 output reuses x (x dead by then)
    float* pooled = ws + BUF;           // 16 KB
    int*   spans  = (int*)(pooled + B_SZ * D_SZ);

    // f16 region (16B-aligned: all prior offsets are multiples of 4 floats)
    _Float16* hb   = (_Float16*)(pooled + B_SZ * D_SZ + 32);
    _Float16* xh   = hb + 0 * BUF;
    _Float16* qh   = hb + 1 * BUF;
    _Float16* kh   = hb + 2 * BUF;
    _Float16* vh   = hb + 3 * BUF;
    _Float16* atth = hb + 4 * BUF;
    _Float16* x2h  = hb + 5 * BUF;
    _Float16* hh   = hb + 6 * BUF;
    _Float16* wqh  = hb + 7 * BUF;
    _Float16* wkh  = wqh + WN;
    _Float16* wvh  = wkh + WN;
    _Float16* fcwh = wvh + WN;
    _Float16* f1wh = fcwh + WN;
    _Float16* f2wh = f1wh + WN;

    const int M = B_SZ * N_SZ;            // 16384
    const dim3 gGrid(M / 16, D_SZ / 64);  // (1024, 4)
    const dim3 gBlock(128);
    const int cvtBlocks = (WN + 255) / 256;

    embed_kernel<<<(int)((BUF + 255) / 256), 256, 0, stream>>>(ids, embW, embB, x, xh);
    spans_kernel<<<1, 32, 0, stream>>>(amask, spans);
    cvt_h_kernel<<<cvtBlocks, 256, 0, stream>>>(wq,   wqh,  WN);
    cvt_h_kernel<<<cvtBlocks, 256, 0, stream>>>(wk,   wkh,  WN);
    cvt_h_kernel<<<cvtBlocks, 256, 0, stream>>>(wv,   wvh,  WN);
    cvt_h_kernel<<<cvtBlocks, 256, 0, stream>>>(fcw,  fcwh, WN);
    cvt_h_kernel<<<cvtBlocks, 256, 0, stream>>>(ff1w, f1wh, WN);
    cvt_h_kernel<<<cvtBlocks, 256, 0, stream>>>(ff2w, f2wh, WN);

    gemm_h_kernel<<<gGrid, gBlock, 0, stream>>>(xh, wqh, bq, nullptr, nullptr, qh, M, D_SZ, D_SZ, 0);
    gemm_h_kernel<<<gGrid, gBlock, 0, stream>>>(xh, wkh, bk, nullptr, nullptr, kh, M, D_SZ, D_SZ, 0);
    gemm_h_kernel<<<gGrid, gBlock, 0, stream>>>(xh, wvh, bv, nullptr, nullptr, vh, M, D_SZ, D_SZ, 0);

    attention_kernel<<<B_SZ * H_SZ * (N_SZ / 16), 32, 0, stream>>>(qh, kh, vh, spans, atth);

    // x2 = x + (att @ fc_w^T + fc_b)         (f16 copy for next GEMM)
    gemm_h_kernel<<<gGrid, gBlock, 0, stream>>>(atth, fcwh, fcb, x, nullptr, x2h, M, D_SZ, D_SZ, 0);
    // h = gelu(x2 @ ff1^T + b1)
    gemm_h_kernel<<<gGrid, gBlock, 0, stream>>>(x2h, f1wh, ff1b, nullptr, nullptr, hh, M, HID_SZ, D_SZ, 1);
    // x3 = h @ ff2^T + b2   (f32, feeds mean-pool)
    gemm_h_kernel<<<gGrid, gBlock, 0, stream>>>(hh, f2wh, ff2b, nullptr, x3, nullptr, M, D_SZ, HID_SZ, 0);

    pool_kernel<<<(B_SZ * D_SZ + 255) / 256, 256, 0, stream>>>(x3, pooled);
    head_kernel<<<B_SZ, 128, 0, stream>>>(pooled, h1w, h1b, bng, bnb, bnm, bnv, h2w, h2b,
                                          (float*)d_out);
}
